// MultiHeadedAttention_2585570312467
// MI455X (gfx1250) — matrix operand order repaired and checked by reading, not compiled, not run on hardware
//
#include <hip/hip_runtime.h>

// ---------------------------------------------------------------------------
// MultiHeadedAttention (RoPE, flat reshape, axis-2 softmax) for gfx1250 (MI455X)
// All matmuls via v_wmma_f32_16x16x32_bf16 (wave32 WMMA), f32 accumulation.
// ---------------------------------------------------------------------------

typedef __attribute__((ext_vector_type(16))) __bf16 v16bf;
typedef __attribute__((ext_vector_type(8)))  float  v8f;

#define BATCH 2
#define SEQ   2048
#define DIM   1024
#define NHEAD 16
#define HD    64
#define BH    (BATCH * NHEAD)

#if defined(__has_builtin)
#if __has_builtin(__builtin_amdgcn_cvt_pk_bf16_f32)
#define HAVE_PK_BF16 1
#endif
#endif

union Frag {
    v16bf v;
    uint4 q[2];
};

__device__ __forceinline__ v8f wmma_bf16(v16bf a, v16bf b, v8f c) {
    return __builtin_amdgcn_wmma_f32_16x16x32_bf16(
        /*neg_a=*/false, a, /*neg_b=*/false, b,
        /*c_mod=*/(short)0, c, /*reuse_a=*/false, /*reuse_b=*/false);
}

// pack two f32 -> packed bf16x2 (one v_cvt_pk_bf16_f32 when available)
__device__ __forceinline__ unsigned int pack2(float a, float b) {
#ifdef HAVE_PK_BF16
    auto r = __builtin_amdgcn_cvt_pk_bf16_f32(a, b);
    unsigned int u;
    __builtin_memcpy(&u, &r, sizeof(u));
    return u;
#else
    unsigned int ua = __float_as_uint(a);
    ua += 0x7FFFu + ((ua >> 16) & 1u);
    unsigned int ub = __float_as_uint(b);
    ub += 0x7FFFu + ((ub >> 16) & 1u);
    return (ua >> 16) | (ub & 0xFFFF0000u);
#endif
}

__device__ __forceinline__ unsigned short f2bf(float f) {
#ifdef HAVE_PK_BF16
    return (unsigned short)(pack2(f, f) & 0xFFFFu);
#else
    unsigned int u = __float_as_uint(f);
    u += 0x7FFFu + ((u >> 16) & 1u);   // round-to-nearest-even
    return (unsigned short)(u >> 16);
#endif
}

// ---------------------------------------------------------------------------
// Generic GEMM: C[M,1024] = A[M,1024] @ W[1024,1024]^T + bias   (f32 in/out)
// WG = 256 threads = 8 waves, block tile 64(M) x 128(N), wave tile 32x32.
// grid = (N/128, M/64)
// ---------------------------------------------------------------------------
__global__ __launch_bounds__(256) void gemm_bf16_wt(
    const float* __restrict__ A, const float* __restrict__ W,
    const float* __restrict__ bias, float* __restrict__ C) {

    __shared__ unsigned short Asm[64 * 32];    // A tile  (M x Kc) row-major
    __shared__ unsigned short Wsm[128 * 32];   // W tile  (N x Kc) row-major (== B^T rows)

    const int wid  = threadIdx.x >> 5;
    const int lane = threadIdx.x & 31;
    const int n    = lane & 15;
    const int hi   = lane >> 4;            // 0 or 1
    const int wm   = wid & 1;              // wave row  (0..1)
    const int wn   = wid >> 1;             // wave col  (0..3)
    const int mbase = blockIdx.y * 64;
    const int nbase = blockIdx.x * 128;

    // per-thread staging coordinates
    const int arow = threadIdx.x >> 2, ac0 = (threadIdx.x & 3) * 8;
    const int wrow = threadIdx.x >> 1, wc0 = (threadIdx.x & 1) * 16;
    const float* aSrc = A + (size_t)(mbase + arow) * 1024 + ac0;
    const float* wSrc = W + (size_t)(nbase + wrow) * 1024 + wc0;

    v8f acc[2][2];
    #pragma unroll
    for (int i = 0; i < 2; ++i)
        #pragma unroll
        for (int j = 0; j < 2; ++j)
            acc[i][j] = v8f{0.f, 0.f, 0.f, 0.f, 0.f, 0.f, 0.f, 0.f};

    for (int kb = 0; kb < 1024; kb += 32) {
        __syncthreads();
        {   // stage A tile: 64 rows x 32 halves; thread -> 8 halves
            const float* s = aSrc + kb;
            float4 f0 = *(const float4*)s;
            float4 f1 = *(const float4*)(s + 4);
            *(uint4*)&Asm[arow * 32 + ac0] =
                make_uint4(pack2(f0.x, f0.y), pack2(f0.z, f0.w),
                           pack2(f1.x, f1.y), pack2(f1.z, f1.w));
        }
        {   // stage W tile: 128 rows x 32 halves; thread -> 16 halves
            const float* s = wSrc + kb;
            float4 f0 = *(const float4*)s;
            float4 f1 = *(const float4*)(s + 4);
            float4 f2 = *(const float4*)(s + 8);
            float4 f3 = *(const float4*)(s + 12);
            unsigned short* d = &Wsm[wrow * 32 + wc0];
            *(uint4*)d       = make_uint4(pack2(f0.x, f0.y), pack2(f0.z, f0.w),
                                          pack2(f1.x, f1.y), pack2(f1.z, f1.w));
            *(uint4*)(d + 8) = make_uint4(pack2(f2.x, f2.y), pack2(f2.z, f2.w),
                                          pack2(f3.x, f3.y), pack2(f3.z, f3.w));
        }
        // pull next K-chunk toward the caches while we compute (global_prefetch_b8)
        if (kb + 32 < 1024) {
            __builtin_prefetch(aSrc + kb + 32, 0, 1);
            __builtin_prefetch(wSrc + kb + 32, 0, 1);
        }
        __syncthreads();

        Frag af[2], bf[2];
        #pragma unroll
        for (int i = 0; i < 2; ++i) {   // A fragment: row = local m, runs at hi*8 and hi*8+16
            const unsigned short* p = &Asm[(wm * 32 + i * 16 + n) * 32 + hi * 8];
            af[i].q[0] = *(const uint4*)p;
            af[i].q[1] = *(const uint4*)(p + 16);
        }
        #pragma unroll
        for (int j = 0; j < 2; ++j) {   // B fragment: col = n, runs at hi*8 and hi*8+16
            const unsigned short* p = &Wsm[(wn * 32 + j * 16 + n) * 32 + hi * 8];
            bf[j].q[0] = *(const uint4*)p;
            bf[j].q[1] = *(const uint4*)(p + 16);
        }
        #pragma unroll
        for (int i = 0; i < 2; ++i)
            #pragma unroll
            for (int j = 0; j < 2; ++j)
                acc[i][j] = wmma_bf16(af[i].v, bf[j].v, acc[i][j]);
    }

    #pragma unroll
    for (int i = 0; i < 2; ++i)
        #pragma unroll
        for (int j = 0; j < 2; ++j) {
            const int col = nbase + wn * 32 + j * 16 + n;
            const float bv = bias[col];
            #pragma unroll
            for (int r = 0; r < 8; ++r) {
                const int row = mbase + wm * 32 + i * 16 + r + hi * 8;
                C[(size_t)row * 1024 + col] = acc[i][j][r] + bv;
            }
        }
}

// ---------------------------------------------------------------------------
// RoPE (flat (B,S,D), reshaped-index semantics) + f32 -> bf16 (+scale)
// ---------------------------------------------------------------------------
__global__ __launch_bounds__(256) void rope_to_bf16(
    const float* __restrict__ src, unsigned short* __restrict__ dst, float outScale) {
    const int i = blockIdx.x * 256 + threadIdx.x;      // < B*S*D
    const int e  = i & 63;
    const int s2 = (i & (SEQ * HD - 1)) >> 6;          // position index in reshaped space
    const int j  = e & 31;
    const float inv = __expf(-(float)j * 0.28782360963f);  // ln(10000)/32
    float sn, cs;
    __sincosf((float)s2 * inv, &sn, &cs);
    const float x  = src[i];
    const float xp = src[(e < 32) ? (i + 32) : (i - 32)];
    const float r  = x * cs + ((e < 32) ? -xp : xp) * sn;
    dst[i] = f2bf(r * outScale);
}

// ---------------------------------------------------------------------------
// RoPE for V + transpose to [bh][d][k] (k contiguous) so pass-2 B-fragments
// are contiguous global loads. One 64x64 tile per block through LDS.
// grid.x = BH * (SEQ/64)
// ---------------------------------------------------------------------------
__global__ __launch_bounds__(256) void rope_v_transpose(
    const float* __restrict__ src, unsigned short* __restrict__ dstT) {

    __shared__ unsigned short tile[64 * 72];   // [e][s2local], padded stride

    const int bh   = blockIdx.x >> 5;          // 0..31
    const int kblk = blockIdx.x & 31;          // s2 block of 64
    const size_t base = (size_t)bh * (SEQ * HD) + (size_t)kblk * 64 * HD;

    {   // read + rotate: thread -> 16 contiguous elements of one s2 row
        const int r  = threadIdx.x >> 2;       // local s2 row 0..63
        const int e0 = (threadIdx.x & 3) * 16;
        const int s2 = kblk * 64 + r;
        const float* p = src + base + (size_t)r * HD;
        #pragma unroll
        for (int u = 0; u < 16; ++u) {
            const int e = e0 + u;
            const int j = e & 31;
            const float inv = __expf(-(float)j * 0.28782360963f);
            float sn, cs;
            __sincosf((float)s2 * inv, &sn, &cs);
            const float x  = p[e];
            const float xp = p[(e < 32) ? (e + 32) : (e - 32)];
            const float rr = x * cs + ((e < 32) ? -xp : xp) * sn;
            tile[e * 72 + r] = f2bf(rr);
        }
    }
    __syncthreads();
    {   // write transposed: thread -> row d, 16 contiguous k
        const int d  = threadIdx.x >> 2;
        const int k0 = (threadIdx.x & 3) * 16;
        unsigned short t[16];
        #pragma unroll
        for (int u = 0; u < 16; ++u) t[u] = tile[d * 72 + k0 + u];
        unsigned short* o =
            dstT + (size_t)bh * (SEQ * HD) + (size_t)d * SEQ + kblk * 64 + k0;
        *(uint4*)o = make_uint4(
            (unsigned)t[0] | ((unsigned)t[1] << 16), (unsigned)t[2] | ((unsigned)t[3] << 16),
            (unsigned)t[4] | ((unsigned)t[5] << 16), (unsigned)t[6] | ((unsigned)t[7] << 16));
        *(uint4*)(o + 8) = make_uint4(
            (unsigned)t[8]  | ((unsigned)t[9]  << 16), (unsigned)t[10] | ((unsigned)t[11] << 16),
            (unsigned)t[12] | ((unsigned)t[13] << 16), (unsigned)t[14] | ((unsigned)t[15] << 16));
    }
}

// ---------------------------------------------------------------------------
// Pass 1: per-(b,h) column (q-axis) softmax stats.  m[k], rz[k] = 1/sum_q e^{s-m}
// grid = (SEQ/128, BH); each wave owns 16 key columns and streams all q tiles.
// ---------------------------------------------------------------------------
__global__ __launch_bounds__(256) void attn_pass1(
    const unsigned short* __restrict__ qbuf, const unsigned short* __restrict__ kbuf,
    float* __restrict__ mstat, float* __restrict__ rzstat) {

    const int bh   = blockIdx.y;
    const int wid  = threadIdx.x >> 5;
    const int lane = threadIdx.x & 31;
    const int n    = lane & 15;
    const int hi   = lane >> 4;
    const int kbase = blockIdx.x * 128 + wid * 16;

    const unsigned short* Q = qbuf + (size_t)bh * SEQ * HD;
    const unsigned short* K = kbuf + (size_t)bh * SEQ * HD;

    // Resident K^T B-fragments (two 32-wide hd chunks)
    Frag kf[2];
    #pragma unroll
    for (int t = 0; t < 2; ++t) {
        const unsigned short* p = K + (size_t)(kbase + n) * HD + t * 32 + hi * 8;
        kf[t].q[0] = *(const uint4*)p;
        kf[t].q[1] = *(const uint4*)(p + 16);
    }

    const int kg = kbase + n;
    float m = -3.0e38f, Z = 0.f;

    for (int q0 = 0; q0 < SEQ; q0 += 16) {
        v8f sc = v8f{0.f, 0.f, 0.f, 0.f, 0.f, 0.f, 0.f, 0.f};
        #pragma unroll
        for (int t = 0; t < 2; ++t) {
            Frag af;
            const unsigned short* p = Q + (size_t)(q0 + n) * HD + t * 32 + hi * 8;
            af.q[0] = *(const uint4*)p;
            af.q[1] = *(const uint4*)(p + 16);
            sc = wmma_bf16(af.v, kf[t].v, sc);
        }
        float tmax = -3.0e38f;
        #pragma unroll
        for (int r = 0; r < 8; ++r) {
            const int qg = q0 + r + hi * 8;
            const float sv = (qg == kg) ? 0.f : sc[r];   // diag zeroed (1-eye mask)
            sc[r] = sv;
            tmax = fmaxf(tmax, sv);
        }
        const float colm = fmaxf(tmax, __shfl_xor(tmax, 16, 32));
        const float mnew = fmaxf(m, colm);
        float lsum = 0.f;
        #pragma unroll
        for (int r = 0; r < 8; ++r) lsum += __expf(sc[r] - mnew);
        lsum += __shfl_xor(lsum, 16, 32);
        Z = Z * __expf(m - mnew) + lsum;
        m = mnew;
    }
    if (hi == 0) {
        mstat[(size_t)bh * SEQ + kg]  = m;
        rzstat[(size_t)bh * SEQ + kg] = 1.0f / Z;
    }
}

// ---------------------------------------------------------------------------
// Pass 2: out[q,d] = sum_k exp(s[q,k]-m[k])*rz[k] * V[k,d].
// V is pre-transposed [bh][d][k]; probability tile round-trips wave-private LDS
// (DS ops are in-order within a wave -> no block barriers in the loop).
// grid = (SEQ/128, BH); each wave owns 16 q rows (16 x 64 out tile).
// ---------------------------------------------------------------------------
__global__ __launch_bounds__(256) void attn_pass2(
    const unsigned short* __restrict__ qbuf, const unsigned short* __restrict__ kbuf,
    const unsigned short* __restrict__ vtg, const float* __restrict__ mstat,
    const float* __restrict__ rzstat, float* __restrict__ x) {

    __shared__ unsigned short pstage[8 * 16 * 32];  // per-wave probability tiles

    const int bh   = blockIdx.y;
    const int b    = bh >> 4;
    const int h    = bh & 15;
    const int wid  = threadIdx.x >> 5;
    const int lane = threadIdx.x & 31;
    const int n    = lane & 15;
    const int hi   = lane >> 4;
    const int qbase = blockIdx.x * 128 + wid * 16;

    const unsigned short* Q  = qbuf + (size_t)bh * SEQ * HD;
    const unsigned short* K  = kbuf + (size_t)bh * SEQ * HD;
    const unsigned short* Vt = vtg  + (size_t)bh * SEQ * HD;   // [d][k]
    const float* M  = mstat  + (size_t)bh * SEQ;
    const float* RZ = rzstat + (size_t)bh * SEQ;

    // Resident Q A-fragments
    Frag qf[2];
    #pragma unroll
    for (int t = 0; t < 2; ++t) {
        const unsigned short* p = Q + (size_t)(qbase + n) * HD + t * 32 + hi * 8;
        qf[t].q[0] = *(const uint4*)p;
        qf[t].q[1] = *(const uint4*)(p + 16);
    }

    v8f acc[4];
    #pragma unroll
    for (int d = 0; d < 4; ++d) acc[d] = v8f{0.f, 0.f, 0.f, 0.f, 0.f, 0.f, 0.f, 0.f};

    unsigned short* pst = &pstage[wid * 512];

    for (int kb0 = 0; kb0 < SEQ; kb0 += 32) {
        // scores for 2 column tiles of 16 keys -> probabilities staged in LDS
        #pragma unroll
        for (int ct = 0; ct < 2; ++ct) {
            v8f sc = v8f{0.f, 0.f, 0.f, 0.f, 0.f, 0.f, 0.f, 0.f};
            #pragma unroll
            for (int t = 0; t < 2; ++t) {
                Frag kf;
                const unsigned short* p =
                    K + (size_t)(kb0 + ct * 16 + n) * HD + t * 32 + hi * 8;
                kf.q[0] = *(const uint4*)p;
                kf.q[1] = *(const uint4*)(p + 16);
                sc = wmma_bf16(qf[t].v, kf.v, sc);
            }
            const int kg = kb0 + ct * 16 + n;
            const float mv = M[kg];
            const float rz = RZ[kg];
            #pragma unroll
            for (int r = 0; r < 8; ++r) {
                const int qg = qbase + r + hi * 8;
                const float sv = (qg == kg) ? 0.f : sc[r];
                const float pv = __expf(sv - mv) * rz;
                pst[(r + hi * 8) * 32 + ct * 16 + n] = f2bf(pv);
            }
        }
        // P A-fragment (16x32) back from LDS (wave-in-order DS, no barrier)
        Frag pf;
        {
            const unsigned short* p = &pst[n * 32 + hi * 8];
            pf.q[0] = *(const uint4*)p;
            pf.q[1] = *(const uint4*)(p + 16);
        }
        // V B-fragments straight from transposed global V; 4 WMMAs across hd
        #pragma unroll
        for (int dt = 0; dt < 4; ++dt) {
            Frag vf;
            const unsigned short* pv = Vt + (size_t)(dt * 16 + n) * SEQ + kb0 + hi * 8;
            vf.q[0] = *(const uint4*)pv;
            vf.q[1] = *(const uint4*)(pv + 16);
            acc[dt] = wmma_bf16(pf.v, vf.v, acc[dt]);
        }
    }

    // epilogue: x[b, s2, h*64 + d]  (transpose(0,2,1,3).reshape)
    #pragma unroll
    for (int dt = 0; dt < 4; ++dt) {
        const int d = dt * 16 + n;
        #pragma unroll
        for (int r = 0; r < 8; ++r) {
            const int qg = qbase + r + hi * 8;
            x[((size_t)b * SEQ + qg) * DIM + h * 64 + d] = acc[dt][r];
        }
    }
}

// ---------------------------------------------------------------------------
// Launch
// ---------------------------------------------------------------------------
extern "C" void kernel_launch(void* const* d_in, const int* in_sizes, int n_in,
                              void* d_out, int out_size, void* d_ws, size_t ws_size,
                              hipStream_t stream) {
    (void)in_sizes; (void)n_in; (void)out_size; (void)ws_size;

    const float* query = (const float*)d_in[0];
    const float* key   = (const float*)d_in[1];
    const float* value = (const float*)d_in[2];
    /* pos_enc = arange(S), folded analytically */ (void)d_in[3];
    const float* Wk = (const float*)d_in[4];
    const float* bk = (const float*)d_in[5];
    const float* Wv = (const float*)d_in[6];
    const float* bv = (const float*)d_in[7];
    const float* Wo = (const float*)d_in[8];
    const float* bo = (const float*)d_in[9];
    float* out = (float*)d_out;

    char* ws = (char*)d_ws;
    const size_t SZ_PROJ = (size_t)BATCH * SEQ * DIM * 4;   // 16 MB
    const size_t SZ_BF   = (size_t)BATCH * SEQ * DIM * 2;   //  8 MB
    float*          kp  = (float*)(ws);                       // reused as xbuf later
    float*          vp  = (float*)(ws + SZ_PROJ);
    unsigned short* qb  = (unsigned short*)(ws + 2 * SZ_PROJ);
    unsigned short* kbf = (unsigned short*)(ws + 2 * SZ_PROJ + SZ_BF);
    unsigned short* vtg = (unsigned short*)(ws + 2 * SZ_PROJ + 2 * SZ_BF);
    float*          mst = (float*)(ws + 2 * SZ_PROJ + 3 * SZ_BF);
    float*          rzs = (float*)(ws + 2 * SZ_PROJ + 3 * SZ_BF + (size_t)BH * SEQ * 4);
    float*          xb  = kp;   // reuse k-projection buffer after bf16 conversion

    const dim3 gemmGrid(DIM / 128, (BATCH * SEQ) / 64);
    const dim3 attnGrid(SEQ / 128, BH);
    const int ropeBlocks = (BATCH * SEQ * DIM) / 256;

    // 1-2) projections
    gemm_bf16_wt<<<gemmGrid, 256, 0, stream>>>(key,   Wk, bk, kp);
    gemm_bf16_wt<<<gemmGrid, 256, 0, stream>>>(value, Wv, bv, vp);
    // 3) RoPE + bf16 (1/sqrt(hd) folded into q); V also rotated + transposed
    rope_to_bf16<<<ropeBlocks, 256, 0, stream>>>(query, qb, 0.125f);
    rope_to_bf16<<<ropeBlocks, 256, 0, stream>>>(kp,    kbf, 1.0f);
    rope_v_transpose<<<BH * (SEQ / 64), 256, 0, stream>>>(vp, vtg);
    // 4) q-axis softmax stats
    attn_pass1<<<attnGrid, 256, 0, stream>>>(qb, kbf, mst, rzs);
    // 5) attention output (into reused buffer, (b,s,h*64+d) layout)
    attn_pass2<<<attnGrid, 256, 0, stream>>>(qb, kbf, vtg, mst, rzs, xb);
    // 6) output projection
    gemm_bf16_wt<<<gemmGrid, 256, 0, stream>>>(xb, Wo, bo, out);
}
